// CrossAttention_3307124818375
// MI455X (gfx1250) — matrix operand order repaired and checked by reading, not compiled, not run on hardware
//
#include <hip/hip_runtime.h>

typedef __attribute__((ext_vector_type(16))) _Float16 v16h;
typedef __attribute__((ext_vector_type(8)))  float    v8f;

#define B_  4
#define C_  64
#define N_  4096
#define NT  256      // tiles of 16 along N
#define MJ  128      // chunks of 32 along N (keys)
#define QH_ELEMS (B_*C_*N_)   // 1,048,576 halves per packed tensor (2 MB)

// 16-bit A-operand (16 rows x 32 K) packing: lanes 0-15 hold K=0..15, 16-31 hold K=16..31
__device__ __forceinline__ int a_lane(int r, int k){ return r + 16*((k>>4)&1); }
__device__ __forceinline__ int a_half(int k)       { return k & 15; }
// 16-bit B-operand (32 K x 16 cols) packing: lanes 0-15 hold K=0..15, 16-31 hold K=16..31
__device__ __forceinline__ int b_lane(int n, int k){ return n + 16*((k>>4)&1); }
__device__ __forceinline__ int b_half(int k)       { return k & 15; }

// -------- Kernel 1: QKV 1x1-conv projections, convert to f16, store in WMMA
// operand layouts.  Q gets the 1/sqrt(C) scale folded in and is stored as
// B-operands; K and V are stored as A-operands.
__global__ __launch_bounds__(256) void proj_pack_kernel(
    const float* __restrict__ query, const float* __restrict__ support,
    const float* __restrict__ Wq, const float* __restrict__ bq,
    const float* __restrict__ Wk, const float* __restrict__ bk,
    const float* __restrict__ Wv, const float* __restrict__ bv,
    _Float16* __restrict__ Qh, _Float16* __restrict__ Kh, _Float16* __restrict__ Vh)
{
    int gid = blockIdx.x * 256 + threadIdx.x;      // B*C*N = 2^20 threads
    int n = gid & (N_ - 1);
    int o = (gid >> 12) & (C_ - 1);
    int b = gid >> 18;

    const float* qcol = query   + (size_t)b * C_ * N_ + n;
    const float* scol = support + (size_t)b * C_ * N_ + n;
    const float* wq = Wq + o * C_;
    const float* wk = Wk + o * C_;
    const float* wv = Wv + o * C_;

    float aq = 0.f, ak = 0.f, av = 0.f;
#pragma unroll 8
    for (int c = 0; c < C_; ++c) {
        float xq = qcol[(size_t)c * N_];
        float xs = scol[(size_t)c * N_];
        aq = fmaf(wq[c], xq, aq);
        ak = fmaf(wk[c], xs, ak);
        av = fmaf(wv[c], xs, av);
    }
    aq = (aq + bq[o]) * 0.125f;   // fold 1/sqrt(64) into Q
    ak += bk[o];
    av += bv[o];

    // Q: B-layout tiles [b][ntile][cchunk]; element (k=o&31, col=n&15)
    {
        int t = n >> 4, ncol = n & 15, cc = o >> 5, k = o & 31;
        size_t base = ((size_t)((b * NT + t) * 2 + cc)) * 512;
        Qh[base + b_lane(ncol, k) * 16 + b_half(k)] = (_Float16)aq;
    }
    // K: A-layout tiles [b][mtile][cchunk]; element (row=n&15, k=o&31)
    {
        int t = n >> 4, r = n & 15, cc = o >> 5, k = o & 31;
        size_t base = ((size_t)((b * NT + t) * 2 + cc)) * 512;
        Kh[base + a_lane(r, k) * 16 + a_half(k)] = (_Float16)ak;
    }
    // V: A-layout tiles [b][ctile][mchunk]; element (row=o&15, k=n&31)
    {
        int ct = o >> 4, r = o & 15, j = n >> 5, k = n & 31;
        size_t base = ((size_t)((b * 4 + ct) * MJ + j)) * 512;
        Vh[base + a_lane(r, k) * 16 + a_half(k)] = (_Float16)av;
    }
}

// -------- Kernel 2: fused flash attention.  One wave per (batch, 16-query
// tile).  Computes S^T tiles = K_tile^T Q (so softmax rows live inside a lane
// pair), online softmax, then O^T += V * P^T, all in registers via WMMA.
__global__ __launch_bounds__(128) void flash_attn_kernel(
    const _Float16* __restrict__ Qh, const _Float16* __restrict__ Kh,
    const _Float16* __restrict__ Vh, const float* __restrict__ query,
    float* __restrict__ out)
{
    const int lane = threadIdx.x & 31;
    const int wave = blockIdx.x * (blockDim.x >> 5) + (threadIdx.x >> 5);
    const int b = wave >> 8;           // / NT
    const int t = wave & (NT - 1);     // query tile
    const bool lo = lane < 16;

    // Q B-operands for both 32-wide c-chunks (kept in registers all loop long)
    const _Float16* qbase = Qh + ((size_t)((b * NT + t) * 2)) * 512 + lane * 16;
    v16h qb0 = *(const v16h*)(qbase);
    v16h qb1 = *(const v16h*)(qbase + 512);

    v8f acc0 = {}, acc1 = {}, acc2 = {}, acc3 = {};   // O^T, 64 c-rows
    float run_max = -1.0e30f, run_sum = 0.f;

    for (int j = 0; j < MJ; ++j) {
        const _Float16* kbase = Kh + ((size_t)((b * NT + 2 * j) * 2)) * 512 + lane * 16;
        v16h ka00 = *(const v16h*)(kbase);            // m-tile 2j,   c-chunk 0
        v16h ka01 = *(const v16h*)(kbase + 512);      // m-tile 2j,   c-chunk 1
        v16h ka10 = *(const v16h*)(kbase + 1024);     // m-tile 2j+1, c-chunk 0
        v16h ka11 = *(const v16h*)(kbase + 1536);     // m-tile 2j+1, c-chunk 1
        const _Float16* vbase = Vh + ((size_t)(b * 4) * MJ + j) * 512 + lane * 16;
        v16h va0 = *(const v16h*)(vbase);
        v16h va1 = *(const v16h*)(vbase + (size_t)MJ * 512);
        v16h va2 = *(const v16h*)(vbase + (size_t)2 * MJ * 512);
        v16h va3 = *(const v16h*)(vbase + (size_t)3 * MJ * 512);
        if (j + 1 < MJ) {
            __builtin_prefetch((const void*)(kbase + 2048), 0, 1);
            __builtin_prefetch((const void*)(vbase + 512), 0, 1);
        }

        // S^T tiles: D[m_local, n] = sum_c K[c,m] * Q[c,n]  (scale folded in Q)
        v8f s0 = {}, s1 = {};
        s0 = __builtin_amdgcn_wmma_f32_16x16x32_f16(false, ka00, false, qb0, (short)0, s0, false, false);
        s0 = __builtin_amdgcn_wmma_f32_16x16x32_f16(false, ka01, false, qb1, (short)0, s0, false, false);
        s1 = __builtin_amdgcn_wmma_f32_16x16x32_f16(false, ka10, false, qb0, (short)0, s1, false, false);
        s1 = __builtin_amdgcn_wmma_f32_16x16x32_f16(false, ka11, false, qb1, (short)0, s1, false, false);

        // Online softmax per query row n = lane&15 (lane pair l, l^16 shares n)
        float tmax = s0[0];
#pragma unroll
        for (int g = 1; g < 8; ++g) tmax = fmaxf(tmax, s0[g]);
#pragma unroll
        for (int g = 0; g < 8; ++g) tmax = fmaxf(tmax, s1[g]);
        tmax = fmaxf(tmax, __shfl_xor(tmax, 16, 32));
        float nmax = fmaxf(run_max, tmax);
        float corr = __expf(run_max - nmax);
        float p0[8], p1[8], tsum = 0.f;
#pragma unroll
        for (int g = 0; g < 8; ++g) {
            p0[g] = __expf(s0[g] - nmax);
            p1[g] = __expf(s1[g] - nmax);
            tsum += p0[g] + p1[g];
        }
        tsum += __shfl_xor(tsum, 16, 32);
        run_sum = run_sum * corr + tsum;
        run_max = nmax;
#pragma unroll
        for (int g = 0; g < 8; ++g) {
            acc0[g] *= corr; acc1[g] *= corr; acc2[g] *= corr; acc3[g] *= corr;
        }

        // Pack P^T (32 m x 16 n) into a B-operand: lanes 0-15 need m=0..15,
        // lanes 16-31 need m=16..31 -> one xor-16 exchange, no LDS.
        v16h pb;
#pragma unroll
        for (int g = 0; g < 8; ++g) {
            float a  = p0[g], bb = p1[g];
            float ax = __shfl_xor(a, 16, 32);
            float bx = __shfl_xor(bb, 16, 32);
            pb[g]     = (_Float16)(lo ? a  : bx);
            pb[g + 8] = (_Float16)(lo ? ax : bb);
        }

        // O^T[c, n] += sum_m V[c, m] * P[n, m]
        acc0 = __builtin_amdgcn_wmma_f32_16x16x32_f16(false, va0, false, pb, (short)0, acc0, false, false);
        acc1 = __builtin_amdgcn_wmma_f32_16x16x32_f16(false, va1, false, pb, (short)0, acc1, false, false);
        acc2 = __builtin_amdgcn_wmma_f32_16x16x32_f16(false, va2, false, pb, (short)0, acc2, false, false);
        acc3 = __builtin_amdgcn_wmma_f32_16x16x32_f16(false, va3, false, pb, (short)0, acc3, false, false);
    }

    float inv = 1.f / run_sum;
    int n   = t * 16 + (lane & 15);
    int chi = (lane >> 4) * 8;
    v8f accs[4] = {acc0, acc1, acc2, acc3};
#pragma unroll
    for (int ct = 0; ct < 4; ++ct) {
#pragma unroll
        for (int g = 0; g < 8; ++g) {
            int c = ct * 16 + chi + g;
            size_t idx = ((size_t)(b * C_ + c)) * N_ + n;
            out[idx] = accs[ct][g] * inv + query[idx];   // residual add
        }
    }
}

extern "C" void kernel_launch(void* const* d_in, const int* in_sizes, int n_in,
                              void* d_out, int out_size, void* d_ws, size_t ws_size,
                              hipStream_t stream) {
    (void)in_sizes; (void)n_in; (void)out_size; (void)ws_size;
    const float* query   = (const float*)d_in[0];
    const float* support = (const float*)d_in[1];
    const float* Wq = (const float*)d_in[2];
    const float* bq = (const float*)d_in[3];
    const float* Wk = (const float*)d_in[4];
    const float* bk = (const float*)d_in[5];
    const float* Wv = (const float*)d_in[6];
    const float* bv = (const float*)d_in[7];

    _Float16* Qh = (_Float16*)d_ws;        // 2 MB
    _Float16* Kh = Qh + QH_ELEMS;          // 2 MB
    _Float16* Vh = Kh + QH_ELEMS;          // 2 MB

    proj_pack_kernel<<<(B_ * C_ * N_) / 256, 256, 0, stream>>>(
        query, support, Wq, bq, Wk, bk, Wv, bv, Qh, Kh, Vh);

    // 1024 waves = B * (N/16) query tiles; 4 waves per block
    flash_attn_kernel<<<(B_ * NT) / 4, 128, 0, stream>>>(Qh, Kh, Vh, query, (float*)d_out);
}
